// AttnBlock_24550033064216
// MI455X (gfx1250) — matrix operand order repaired and checked by reading, not compiled, not run on hardware
//
#include <hip/hip_runtime.h>

typedef __attribute__((ext_vector_type(16))) _Float16 v16h;
typedef __attribute__((ext_vector_type(8)))  _Float16 v8h;
typedef __attribute__((ext_vector_type(8)))  float    v8f;

#define DEV __device__ static __forceinline__

constexpr int BB  = 4;
constexpr int C   = 512;
constexpr int HW  = 4096;       // 64*64 tokens
constexpr int GRP = 32;
constexpr int CPG = C / GRP;    // 16 channels per group
constexpr float SM_SCALE = 0.04419417382415922f;  // C^-0.5

DEV int laneid() { return (int)(threadIdx.x & 31); }

// A-fragment: 16x32 f16 tile, row-major source with row stride ld (halves).
// lane L: row = L&15 ; elem h -> K = (h>=8)*16 + (L>=16)*8 + (h&7)
DEV v16h frag_a(const _Float16* p, int ld) {
  const int l = laneid();
  const _Float16* r = p + (size_t)(l & 15) * ld + ((l >> 4) << 3);
  v8h lo = *(const v8h*)(r);        // K = koff .. koff+7
  v8h hi = *(const v8h*)(r + 16);   // K = 16+koff .. 16+koff+7
  return __builtin_shufflevector(lo, hi, 0,1,2,3,4,5,6,7,8,9,10,11,12,13,14,15);
}

// B-fragment: 32x16 (KxN) fed from B^T row-major (N rows, K cols), stride ld.
// lane L: col N = L&15 ; elem h -> K = (h>=8)*16 + (L>=16)*8 + (h&7)
DEV v16h frag_b(const _Float16* p, int ld) {
  const int l = laneid();
  const _Float16* r = p + (size_t)(l & 15) * ld + ((l >> 4) << 3);
  v8h lo = *(const v8h*)(r);
  v8h hi = *(const v8h*)(r + 16);
  return __builtin_shufflevector(lo, hi, 0,1,2,3,4,5,6,7,8,9,10,11,12,13,14,15);
}

DEV v8f wmma_f16(v16h a, v16h b, v8f c) {
  return __builtin_amdgcn_wmma_f32_16x16x32_f16(false, a, false, b, (short)0, c,
                                                false, false);
}

DEV v8f vzero8() { v8f z = {0.f,0.f,0.f,0.f,0.f,0.f,0.f,0.f}; return z; }

// gfx1250 async copy: global -> LDS, no VGPR data movement, tracked by ASYNCcnt
DEV void async_ld_b128(unsigned int lds_off, const void* g) {
  asm volatile("global_load_async_to_lds_b128 %0, %1, off"
               :: "v"(lds_off), "v"(g)
               : "memory");
}
DEV void wait_async0() { asm volatile("s_wait_asynccnt 0x0" ::: "memory"); }

// ---------------------------------------------------------------------------
// Kernel 1: f32 -> f16 weight conversion (4 matrices of 512x512)
// ---------------------------------------------------------------------------
__global__ void wcvt_kernel(const float* a, const float* b, const float* c,
                            const float* d, _Float16* A, _Float16* B,
                            _Float16* Cc, _Float16* D) {
  int i = blockIdx.x * blockDim.x + threadIdx.x;
  if (i < C * C) {
    A[i]  = (_Float16)a[i];
    B[i]  = (_Float16)b[i];
    Cc[i] = (_Float16)c[i];
    D[i]  = (_Float16)d[i];
  }
}

// ---------------------------------------------------------------------------
// Kernel 2: GroupNorm. One block per (batch, group). Writes h as (B, N, C) f16
// (token-major => directly consumable as WMMA A-fragments).
// ---------------------------------------------------------------------------
__global__ __launch_bounds__(256) void gn_kernel(const float* __restrict__ x,
                                                 const float* __restrict__ gw,
                                                 const float* __restrict__ gb,
                                                 _Float16* __restrict__ h) {
  const int bg = blockIdx.x;
  const int b  = bg / GRP;
  const int g  = bg % GRP;
  const int tid = threadIdx.x;
  const float* xg = x + ((size_t)(b * C + g * CPG)) * HW;  // 65536 contiguous

  float s = 0.f, s2 = 0.f;
  for (int i = tid * 4; i < CPG * HW; i += 256 * 4) {
    float4 v = *(const float4*)(xg + i);
    s  += v.x + v.y + v.z + v.w;
    s2 += v.x * v.x + v.y * v.y + v.z * v.z + v.w * v.w;
  }
  __shared__ float r1[256], r2[256];
  r1[tid] = s; r2[tid] = s2;
  __syncthreads();
  for (int w = 128; w > 0; w >>= 1) {
    if (tid < w) { r1[tid] += r1[tid + w]; r2[tid] += r2[tid + w]; }
    __syncthreads();
  }
  const float inv_n = 1.f / (float)(CPG * HW);
  const float mean = r1[0] * inv_n;
  const float var  = r2[0] * inv_n - mean * mean;
  const float rs   = rsqrtf(var + 1e-5f);

  for (int i = tid * 4; i < CPG * HW; i += 256 * 4) {
    float4 v = *(const float4*)(xg + i);
    int cl = i >> 12;              // i / 4096 -> local channel (same for all 4)
    int n  = i & (HW - 1);
    int cc = g * CPG + cl;
    float wm = gw[cc] * rs;
    float bm = gb[cc] - mean * wm;
    size_t base = ((size_t)(b * HW) + n) * C + cc;
    h[base + 0 * C] = (_Float16)(v.x * wm + bm);
    h[base + 1 * C] = (_Float16)(v.y * wm + bm);
    h[base + 2 * C] = (_Float16)(v.z * wm + bm);
    h[base + 3 * C] = (_Float16)(v.w * wm + bm);
  }
}

// ---------------------------------------------------------------------------
// Kernel 3: fused Q/K/V projection.  q[n,o] = sum_c h[n,c] * W[o,c] + bias.
// Wave = 16 tokens x 64 out-channels, for q, k and v at once (A reuse x12).
// v is stored transposed (B, C, N) so the PV GEMM gets contiguous B-frags.
// ---------------------------------------------------------------------------
__global__ __launch_bounds__(128) void qkv_kernel(
    const _Float16* __restrict__ h, const _Float16* __restrict__ qw,
    const _Float16* __restrict__ kw, const _Float16* __restrict__ vw,
    const float* __restrict__ qb, const float* __restrict__ kb,
    const float* __restrict__ vb, _Float16* __restrict__ q,
    _Float16* __restrict__ k, _Float16* __restrict__ vT) {
  const int t0 = blockIdx.x * 16;
  const int c0 = blockIdx.y * 256 + (threadIdx.x >> 5) * 64;
  const int b  = blockIdx.z;
  const int lane = laneid();

  v8f aq[4], ak[4], av[4];
#pragma unroll
  for (int j = 0; j < 4; ++j) { aq[j] = vzero8(); ak[j] = vzero8(); av[j] = vzero8(); }

  const _Float16* hrow = h + ((size_t)(b * HW + t0)) * C;
#pragma unroll 4
  for (int kc = 0; kc < C / 32; ++kc) {
    v16h a = frag_a(hrow + kc * 32, C);
#pragma unroll
    for (int j = 0; j < 4; ++j) {
      const size_t wb = (size_t)(c0 + j * 16) * C + kc * 32;
      aq[j] = wmma_f16(a, frag_b(qw + wb, C), aq[j]);
      ak[j] = wmma_f16(a, frag_b(kw + wb, C), ak[j]);
      av[j] = wmma_f16(a, frag_b(vw + wb, C), av[j]);
    }
  }

  const int lc = lane & 15;
  const int rb = (lane >> 4) * 8;
#pragma unroll
  for (int j = 0; j < 4; ++j) {
    const int oc = c0 + j * 16 + lc;
    const float bq = qb[oc], bk = kb[oc], bv = vb[oc];
#pragma unroll
    for (int r = 0; r < 8; ++r) {
      const size_t idx = ((size_t)(b * HW + t0 + rb + r)) * C + oc;
      q[idx] = (_Float16)(aq[j][r] + bq);
      k[idx] = (_Float16)(ak[j][r] + bk);
    }
    v8h pk;
#pragma unroll
    for (int r = 0; r < 8; ++r) pk[r] = (_Float16)(av[j][r] + bv);
    *(v8h*)(vT + ((size_t)(b * C + oc)) * HW + t0 + rb) = pk;  // contiguous
  }
}

// ---------------------------------------------------------------------------
// Kernel 4: flash attention.  Block = 256 threads = 8 waves = 4 wave-pairs.
// A wave-pair shares 16 queries; each wave owns 256 output channels
// (16 v8f accumulators).  32-key tiles staged in LDS with
// GLOBAL_LOAD_ASYNC_TO_LDS_B128 (ASYNCcnt); online softmax with half-wave
// shuffle reductions; P re-laid-out through a per-wave LDS buffer.
// ---------------------------------------------------------------------------
__global__ __launch_bounds__(256) void attn_kernel(
    const _Float16* __restrict__ q, const _Float16* __restrict__ k,
    const _Float16* __restrict__ vT, _Float16* __restrict__ o) {
  __shared__ __align__(16) _Float16 sK[32 * C];      // 32 keys x 512 ch (32 KB)
  __shared__ __align__(16) _Float16 sV[C * 32];      // 512 ch x 32 keys (32 KB)
  __shared__ __align__(16) _Float16 sP[8][16 * 32];  // per-wave P tile  (8 KB)

  const int tid  = threadIdx.x;
  const int wave = tid >> 5;
  const int lane = tid & 31;
  const int qg   = wave >> 1;          // query sub-group (0..3)
  const int cbase = (wave & 1) * 256;  // output-channel half
  const int qrow0 = blockIdx.x * 64;
  const int b = blockIdx.y;

  const unsigned int sK_lds = (unsigned int)(uintptr_t)&sK[0];
  const unsigned int sV_lds = (unsigned int)(uintptr_t)&sV[0];

  // Preload this wave's 16-query A-fragments (16 x v16h = 128 VGPRs).
  const _Float16* qrow = q + ((size_t)(b * HW + qrow0 + qg * 16)) * C;
  v16h qf[16];
#pragma unroll
  for (int kc = 0; kc < 16; ++kc) qf[kc] = frag_a(qrow + kc * 32, C);

  v8f oacc[16];
#pragma unroll
  for (int ct = 0; ct < 16; ++ct) oacc[ct] = vzero8();
  v8f mrow, lrow;
#pragma unroll
  for (int r = 0; r < 8; ++r) { mrow[r] = -3.0e38f; lrow[r] = 0.f; }

  const _Float16* kb_ = k + ((size_t)b * HW) * C;
  const _Float16* vb_ = vT + ((size_t)b * C) * HW;
  _Float16* pb = &sP[wave][0];
  const int prow = (lane >> 4) * 8;
  const int pcol = lane & 15;

#pragma unroll 1
  for (int kt = 0; kt < HW / 32; ++kt) {
    // Async-stage K tile (32 rows x 512, contiguous) and V^T tile (512 x 32):
    // direct global -> LDS, no VGPR round trip.
    const _Float16* ksrc = kb_ + (size_t)kt * 32 * C;
    for (int u = tid; u < (32 * C) / 8; u += 256)
      async_ld_b128(sK_lds + (unsigned int)(u * 16), ksrc + u * 8);
    for (int u = tid; u < (C * 32) / 8; u += 256) {
      int row = u >> 2, seg = u & 3;
      async_ld_b128(sV_lds + (unsigned int)(row * 64 + seg * 16),
                    vb_ + (size_t)row * HW + kt * 32 + seg * 8);
    }
    if (kt + 1 < HW / 32)
      __builtin_prefetch(kb_ + (size_t)(kt + 1) * 32 * C + tid * 128, 0, 1);
    wait_async0();
    __syncthreads();

    // S = q K^T (two 16x16 tiles over K=512)
    v8f s0 = vzero8(), s1 = vzero8();
#pragma unroll
    for (int kc = 0; kc < 16; ++kc) {
      s0 = wmma_f16(qf[kc], frag_b(sK + kc * 32, C), s0);
      s1 = wmma_f16(qf[kc], frag_b(sK + 16 * C + kc * 32, C), s1);
    }
    s0 = s0 * SM_SCALE;
    s1 = s1 * SM_SCALE;

    // online softmax: row stats live in one lane's 8 components
    v8f t;
#pragma unroll
    for (int r = 0; r < 8; ++r) t[r] = fmaxf(s0[r], s1[r]);
#pragma unroll
    for (int msk = 1; msk < 16; msk <<= 1)
#pragma unroll
      for (int r = 0; r < 8; ++r) t[r] = fmaxf(t[r], __shfl_xor(t[r], msk, 32));
    v8f mnew, corr;
#pragma unroll
    for (int r = 0; r < 8; ++r) {
      mnew[r] = fmaxf(mrow[r], t[r]);
      corr[r] = __expf(mrow[r] - mnew[r]);
      s0[r] = __expf(s0[r] - mnew[r]);
      s1[r] = __expf(s1[r] - mnew[r]);
    }
    v8f rs;
#pragma unroll
    for (int r = 0; r < 8; ++r) rs[r] = s0[r] + s1[r];
#pragma unroll
    for (int msk = 1; msk < 16; msk <<= 1)
#pragma unroll
      for (int r = 0; r < 8; ++r) rs[r] += __shfl_xor(rs[r], msk, 32);
#pragma unroll
    for (int r = 0; r < 8; ++r) {
      lrow[r] = lrow[r] * corr[r] + rs[r];
      mrow[r] = mnew[r];
    }
#pragma unroll
    for (int ct = 0; ct < 16; ++ct) oacc[ct] = oacc[ct] * corr;

    // P (C/D layout) -> LDS -> A-fragment layout
#pragma unroll
    for (int r = 0; r < 8; ++r) {
      pb[(prow + r) * 32 + pcol]      = (_Float16)s0[r];
      pb[(prow + r) * 32 + 16 + pcol] = (_Float16)s1[r];
    }
    v16h ap = frag_a(pb, 32);

    // O += P * V   (16 channel tiles, K = 32 keys)
#pragma unroll
    for (int ct = 0; ct < 16; ++ct)
      oacc[ct] = wmma_f16(ap, frag_b(sV + (size_t)(cbase + ct * 16) * 32, 32),
                          oacc[ct]);
    __syncthreads();
  }

  // epilogue: divide by l, store f16 (B, N, C)
  v8f linv;
#pragma unroll
  for (int r = 0; r < 8; ++r) linv[r] = 1.f / lrow[r];
  const int m0 = qrow0 + qg * 16 + prow;
  const int c0 = cbase + pcol;
#pragma unroll
  for (int ct = 0; ct < 16; ++ct)
#pragma unroll
    for (int r = 0; r < 8; ++r)
      o[((size_t)(b * HW + m0 + r)) * C + c0 + ct * 16] =
          (_Float16)(oacc[ct][r] * linv[r]);
}

// ---------------------------------------------------------------------------
// Kernel 5: output projection + bias + residual, f32 output in (B, C, N).
// ---------------------------------------------------------------------------
__global__ __launch_bounds__(128) void proj_kernel(
    const _Float16* __restrict__ o, const _Float16* __restrict__ ow,
    const float* __restrict__ ob, const float* __restrict__ x,
    float* __restrict__ out) {
  const int t0 = blockIdx.x * 16;
  const int c0 = blockIdx.y * 256 + (threadIdx.x >> 5) * 64;
  const int b  = blockIdx.z;
  const int lane = laneid();

  v8f acc[4];
#pragma unroll
  for (int j = 0; j < 4; ++j) acc[j] = vzero8();

  const _Float16* orow = o + ((size_t)(b * HW + t0)) * C;
#pragma unroll 4
  for (int kc = 0; kc < C / 32; ++kc) {
    v16h a = frag_a(orow + kc * 32, C);
#pragma unroll
    for (int j = 0; j < 4; ++j)
      acc[j] = wmma_f16(a, frag_b(ow + (size_t)(c0 + j * 16) * C + kc * 32, C),
                        acc[j]);
  }

  const int lc = lane & 15;
  const int rb = (lane >> 4) * 8;
#pragma unroll
  for (int j = 0; j < 4; ++j) {
    const int oc = c0 + j * 16 + lc;
    const float bias = ob[oc];
    const size_t base = ((size_t)(b * C + oc)) * HW + t0 + rb;
    float4 x0 = *(const float4*)(x + base);
    float4 x1 = *(const float4*)(x + base + 4);
    float4 r0, r1;
    r0.x = x0.x + acc[j][0] + bias;  r0.y = x0.y + acc[j][1] + bias;
    r0.z = x0.z + acc[j][2] + bias;  r0.w = x0.w + acc[j][3] + bias;
    r1.x = x1.x + acc[j][4] + bias;  r1.y = x1.y + acc[j][5] + bias;
    r1.z = x1.z + acc[j][6] + bias;  r1.w = x1.w + acc[j][7] + bias;
    *(float4*)(out + base)     = r0;
    *(float4*)(out + base + 4) = r1;
  }
}

// ---------------------------------------------------------------------------
extern "C" void kernel_launch(void* const* d_in, const int* in_sizes, int n_in,
                              void* d_out, int out_size, void* d_ws,
                              size_t ws_size, hipStream_t stream) {
  (void)in_sizes; (void)n_in; (void)out_size; (void)ws_size;
  const float* x   = (const float*)d_in[0];
  const float* gnw = (const float*)d_in[1];
  const float* gnb = (const float*)d_in[2];
  const float* qw  = (const float*)d_in[3];
  const float* qb  = (const float*)d_in[4];
  const float* kw  = (const float*)d_in[5];
  const float* kb  = (const float*)d_in[6];
  const float* vw  = (const float*)d_in[7];
  const float* vb  = (const float*)d_in[8];
  const float* ow  = (const float*)d_in[9];
  const float* ob  = (const float*)d_in[10];
  float* out = (float*)d_out;

  // workspace layout (bytes); each f16 tensor B*HW*C = 16 MB, weights 512 KB
  char* ws = (char*)d_ws;
  _Float16* h   = (_Float16*)(ws + 0);
  _Float16* q   = (_Float16*)(ws + 8388608);
  _Float16* k   = (_Float16*)(ws + 16777216);
  _Float16* vT  = (_Float16*)(ws + 25165824);
  _Float16* o   = (_Float16*)(ws + 33554432);
  _Float16* qwh = (_Float16*)(ws + 41943040);
  _Float16* kwh = (_Float16*)(ws + 42467328);
  _Float16* vwh = (_Float16*)(ws + 42991616);
  _Float16* owh = (_Float16*)(ws + 43515904);

  wcvt_kernel<<<(C * C + 255) / 256, 256, 0, stream>>>(qw, kw, vw, ow, qwh, kwh,
                                                       vwh, owh);
  gn_kernel<<<BB * GRP, 256, 0, stream>>>(x, gnw, gnb, h);
  qkv_kernel<<<dim3(HW / 16, C / 256, BB), 128, 0, stream>>>(
      h, qwh, kwh, vwh, qb, kb, vb, q, k, vT);
  attn_kernel<<<dim3(HW / 64, BB), 256, 0, stream>>>(q, k, vT, o);
  proj_kernel<<<dim3(HW / 16, C / 256, BB), 128, 0, stream>>>(o, owh, ob, x,
                                                              out);
}
